// MultiHeadAttention_44521630990779
// MI455X (gfx1250) — matrix operand order repaired and checked by reading, not compiled, not run on hardware
//
#include <hip/hip_runtime.h>
#include <stdint.h>

#define HIDDEN 512
#define HEADS 8
#define DKK 64
#define BATCH 4
#define SEQ 1024
#define NEGV -1000000000.0f

typedef __attribute__((ext_vector_type(16))) __bf16 v16bf;
typedef __attribute__((ext_vector_type(8)))  float v8f;
typedef __attribute__((ext_vector_type(4)))  unsigned int v4u;
typedef __attribute__((ext_vector_type(8)))  int v8i;
typedef __attribute__((ext_vector_type(4)))  int v4i;

union BFrag { v16bf v; uint4 u[2]; };

__device__ __forceinline__ __bf16 f2bf(float f) { return (__bf16)f; }  // native cvt, RNE

__device__ __forceinline__ v8f v8f_zero() {
    v8f z = {0.f, 0.f, 0.f, 0.f, 0.f, 0.f, 0.f, 0.f};
    return z;
}

__device__ __forceinline__ v8f wmma_bf(v16bf a, v16bf b, v8f c) {
    // D = A(16x32 bf16) * B(32x16 bf16) + C(16x16 f32)
    return __builtin_amdgcn_wmma_f32_16x16x32_bf16(false, a, false, b, (short)0, c, false, false);
}

// Issue a TDM DMA: contiguous `nelem` bf16 elements global -> LDS (1-D tile).
// D# packing per CDNA5 ISA.
// Toolchain here is the 6-arg (clang-23/therock) form:
//   (uint32x4 g0, int32x8 g1, int32x4 g2, int32x4 g3, int32x8 pad, i32 cpol)
__device__ __forceinline__ void tdm_load_1d(const __bf16* gsrc, uint32_t lds_off,
                                            unsigned nelem) {
    const uint64_t ga = (uint64_t)(size_t)gsrc;
    v4u g0;
    g0[0] = 1u;                                        // count=1, user descriptor
    g0[1] = lds_off;                                   // lds_addr
    g0[2] = (uint32_t)ga;                              // global_addr[31:0]
    g0[3] = (uint32_t)((ga >> 32) & 0x01FFFFFFu)       // global_addr[56:32]
          | 0x80000000u;                               // type=2 ("image")
    v8i g1;
    g1[0] = (int)0x00010000;                           // data_size=1 (2 bytes)
    g1[1] = (int)((nelem & 0xFFFFu) << 16);            // tensor_dim0[15:0]
    g1[2] = (int)((nelem >> 16) | (1u << 16));         // tensor_dim0[31:16] | tensor_dim1=1
    g1[3] = (int)((nelem & 0xFFFFu) << 16);            // tile_dim0 = nelem
    g1[4] = 0;                                         // tile_dim1=0 (unused), tile_dim2=0
    g1[5] = (int)nelem;                                // tensor_dim0_stride[31:0]
    g1[6] = 0;
    g1[7] = 0;
    v4i gz4 = {0, 0, 0, 0};                            // groups 2/3 unused (<=2D)
    v8i gz8 = {0, 0, 0, 0, 0, 0, 0, 0};
    __builtin_amdgcn_tensor_load_to_lds(g0, g1, gz4, gz4, gz8, 0);
}

// ---------------------------------------------------------------------------
// Tiled GEMM:  out[m,n] = (sum_k A[m,k] * W[n,k] + bias[n]) * alpha
// Block tile 128x128, K-step 32, 8 waves (4x2), wave tile 32x64 (2x4 frags).
// A_BF16:  A is bf16 [M,512], staged with global_load_async_to_lds_b128;
//          otherwise f32 [M,512] converted to bf16 while staging.
// OUT_F32: plain f32 [M,512] output; otherwise bf16 head-interleaved
//          [B, heads, S, dk] output (feeds the attention kernel).
// ---------------------------------------------------------------------------
template <bool A_BF16, bool OUT_F32>
__global__ __launch_bounds__(256)
void gemm_wmma(const void* __restrict__ AP, const float* __restrict__ W,
               const float* __restrict__ bias, void* __restrict__ outP, float alpha)
{
    __shared__ __bf16 As[128 * 32];   // [m][k]
    __shared__ __bf16 Bs[128 * 32];   // [n][k]  (n-major == B-operand layout)

    const int tid  = threadIdx.x;
    const int lane = tid & 31, wave = tid >> 5;
    const int l15  = lane & 15, hi = lane >> 4;
    const int wm   = wave >> 1, wn = wave & 1;
    const int m0   = blockIdx.y * 128, n0 = blockIdx.x * 128;

    v8f acc[2][4];
#pragma unroll
    for (int i = 0; i < 2; ++i)
#pragma unroll
        for (int j = 0; j < 4; ++j) acc[i][j] = v8f_zero();

    for (int kk = 0; kk < HIDDEN; kk += 32) {
        // ---- stage A tile ----
        if constexpr (A_BF16) {
            const __bf16* A = (const __bf16*)AP;
#pragma unroll
            for (int i = 0; i < 2; ++i) {
                int lin = tid + 256 * i;           // 512 chunks of 8 halfs
                int row = lin >> 2, c8 = lin & 3;
                uint32_t ldsoff = (uint32_t)(size_t)(&As[row * 32 + c8 * 8]);
                uint64_t ga = (uint64_t)(size_t)(A + (size_t)(m0 + row) * HIDDEN
                                                 + kk + c8 * 8);
                // async global -> LDS copy (ASYNCcnt path)
                asm volatile("global_load_async_to_lds_b128 %0, %1, off"
                             :: "v"(ldsoff), "v"(ga) : "memory");
            }
        } else {
            const float* A = (const float*)AP;
#pragma unroll
            for (int i = 0; i < 4; ++i) {
                int lin = tid + 256 * i;           // 1024 chunks of 4 floats
                int row = lin >> 3, c4 = lin & 7;
                float4 d = *(const float4*)(A + (size_t)(m0 + row) * HIDDEN + kk + c4 * 4);
                union { __bf16 h[4]; uint2 u; } cv;
                cv.h[0] = f2bf(d.x); cv.h[1] = f2bf(d.y);
                cv.h[2] = f2bf(d.z); cv.h[3] = f2bf(d.w);
                *(uint2*)(&As[row * 32 + c4 * 4]) = cv.u;
            }
        }
        // ---- stage B tile from W (f32, [n][k] row-major already) ----
#pragma unroll
        for (int i = 0; i < 4; ++i) {
            int lin = tid + 256 * i;
            int row = lin >> 3, c4 = lin & 7;
            float4 d = *(const float4*)(W + (size_t)(n0 + row) * HIDDEN + kk + c4 * 4);
            union { __bf16 h[4]; uint2 u; } cv;
            cv.h[0] = f2bf(d.x); cv.h[1] = f2bf(d.y);
            cv.h[2] = f2bf(d.z); cv.h[3] = f2bf(d.w);
            *(uint2*)(&Bs[row * 32 + c4 * 4]) = cv.u;
        }
        if constexpr (A_BF16) {
            asm volatile("s_wait_asynccnt 0" ::: "memory");
        }
        __syncthreads();

        // ---- fragment loads + WMMA ----
        BFrag af[2];
#pragma unroll
        for (int i = 0; i < 2; ++i) {
            const __bf16* p = &As[(wm * 32 + i * 16 + l15) * 32];
            af[i].u[0] = *(const uint4*)(p + hi * 8);        // K = hi*8 + 0..7
            af[i].u[1] = *(const uint4*)(p + 16 + hi * 8);   // K = 16 + hi*8 + 0..7
        }
#pragma unroll
        for (int j = 0; j < 4; ++j) {
            const __bf16* p = &Bs[(wn * 64 + j * 16 + l15) * 32];
            BFrag bf;
            bf.u[0] = *(const uint4*)(p + hi * 8);
            bf.u[1] = *(const uint4*)(p + 16 + hi * 8);
#pragma unroll
            for (int i = 0; i < 2; ++i)
                acc[i][j] = wmma_bf(af[i].v, bf.v, acc[i][j]);
        }
        __syncthreads();
    }

    // ---- epilogue ----
#pragma unroll
    for (int j = 0; j < 4; ++j) {
        const int n  = n0 + wn * 64 + j * 16 + l15;
        const float bn = bias[n];
#pragma unroll
        for (int i = 0; i < 2; ++i) {
#pragma unroll
            for (int r = 0; r < 8; ++r) {
                const int m = m0 + wm * 32 + i * 16 + hi * 8 + r;
                const float val = (acc[i][j][r] + bn) * alpha;
                if constexpr (OUT_F32) {
                    ((float*)outP)[(size_t)m * HIDDEN + n] = val;
                } else {
                    const int b_ = m >> 10, s_ = m & 1023;
                    const int h_ = n >> 6,  d_ = n & 63;
                    ((__bf16*)outP)[(((size_t)b_ * HEADS + h_) * SEQ + s_) * DKK + d_] =
                        f2bf(val);
                }
            }
        }
    }
}

// ---------------------------------------------------------------------------
// Flash attention over [B, heads] with masked-bias softmax.
// Grid (S/64, heads, B), 128 threads = 4 waves; each wave owns 16 q-rows.
// K tile staged via the Tensor Data Mover; V tile transposed through VGPRs.
// ---------------------------------------------------------------------------
__global__ __launch_bounds__(128)
void flash_attn(const __bf16* __restrict__ qh, const __bf16* __restrict__ kh,
                const __bf16* __restrict__ vh, const float* __restrict__ attn_bias,
                const int* __restrict__ gmask, const float* __restrict__ subsq,
                __bf16* __restrict__ aw)
{
    __shared__ __bf16 Ks[64 * 64];      // [kv][dk]  (B-operand for Q*K^T)
    __shared__ __bf16 Vt[64 * 64];      // [dk][kv]  (B-operand for P*V)
    __shared__ __bf16 Ps[4][16 * 64];   // per-wave P staging (C->A relayout)

    const int tid  = threadIdx.x;
    const int lane = tid & 31, wave = tid >> 5;
    const int l15  = lane & 15, hi = lane >> 4;
    const int h = blockIdx.y, b = blockIdx.z;
    const int bh = b * HEADS + h;
    const int q0 = blockIdx.x * 64 + wave * 16;

    // preload Q as two 16x32 A-fragments (K-dim = dk)
    BFrag qf[2];
    {
        const __bf16* qp = qh + ((size_t)bh * SEQ + q0 + l15) * DKK;
        qf[0].u[0] = *(const uint4*)(qp + hi * 8);
        qf[0].u[1] = *(const uint4*)(qp + 16 + hi * 8);
        qf[1].u[0] = *(const uint4*)(qp + 32 + hi * 8);
        qf[1].u[1] = *(const uint4*)(qp + 48 + hi * 8);
    }

    float rm[8], rl[8];
#pragma unroll
    for (int r = 0; r < 8; ++r) { rm[r] = -3.0e38f; rl[r] = 0.0f; }
    v8f o[4];
#pragma unroll
    for (int j = 0; j < 4; ++j) o[j] = v8f_zero();

    const __bf16* kbase = kh + (size_t)bh * SEQ * DKK;
    const __bf16* vbase = vh + (size_t)bh * SEQ * DKK;
    const uint32_t ks_off = (uint32_t)(size_t)(&Ks[0]);

    for (int t = 0; t < SEQ / 64; ++t) {
        const int kv0 = t * 64;
        const __bf16* kp = kbase + (size_t)kv0 * DKK;
        const __bf16* vp = vbase + (size_t)kv0 * DKK;

        // ---- K tile: Tensor Data Mover (one wave issues the DMA) ----
        if (wave == 0) {
            tdm_load_1d(kp, ks_off, 64 * DKK);   // 8 KB contiguous
        }
        // ---- V tile: load row-major, store transposed [dk][kv] ----
#pragma unroll
        for (int i = 0; i < 4; ++i) {
            int lin = tid + 128 * i;            // 512 chunks of 8 halfs
            int row = lin >> 3, c8 = lin & 7;
            union { uint4 u; __bf16 hh[8]; } vv;
            vv.u = *(const uint4*)(vp + row * 64 + c8 * 8);
#pragma unroll
            for (int e = 0; e < 8; ++e)
                Vt[(c8 * 8 + e) * 64 + row] = vv.hh[e];
        }
        if (t + 1 < SEQ / 64) {                 // gfx1250 global_prefetch_b8
            __builtin_prefetch(vp + (size_t)64 * DKK + tid * 32, 0, 1);
        }
        if (wave == 0) {
            __builtin_amdgcn_s_wait_tensorcnt(0);
        }
        __syncthreads();

        // ---- S = Q * K^T (scale folded into Q projection) ----
        v8f s[4];
#pragma unroll
        for (int j = 0; j < 4; ++j) {
            s[j] = v8f_zero();
            const __bf16* p = &Ks[(j * 16 + l15) * 64 + hi * 8];
            BFrag b0, b1;
            b0.u[0] = *(const uint4*)(p);       b0.u[1] = *(const uint4*)(p + 16);
            b1.u[0] = *(const uint4*)(p + 32);  b1.u[1] = *(const uint4*)(p + 48);
            s[j] = wmma_bf(qf[0].v, b0.v, s[j]);
            s[j] = wmma_bf(qf[1].v, b1.v, s[j]);
        }

        // ---- bias + graph mask + subsequent mask ----
#pragma unroll
        for (int j = 0; j < 4; ++j) {
            const int nc = kv0 + j * 16 + l15;
#pragma unroll
            for (int r = 0; r < 8; ++r) {
                const int mr = q0 + hi * 8 + r;
                const size_t bi = ((size_t)bh * SEQ + mr) * SEQ + nc;
                const size_t mi = ((size_t)b  * SEQ + mr) * SEQ + nc;
                const float x = gmask[mi]
                    ? (s[j][r] + attn_bias[bi] + subsq[mi]) : NEGV;
                s[j][r] = x;
            }
        }

        // ---- online softmax (row = VGPR index; reduce across 16 lanes) ----
#pragma unroll
        for (int r = 0; r < 8; ++r) {
            float tm = fmaxf(fmaxf(s[0][r], s[1][r]), fmaxf(s[2][r], s[3][r]));
#pragma unroll
            for (int off = 1; off < 16; off <<= 1)
                tm = fmaxf(tm, __shfl_xor(tm, off, 32));
            const float mnew = fmaxf(rm[r], tm);
            const float al   = __expf(rm[r] - mnew);
            rm[r] = mnew;
            float rs = 0.0f;
#pragma unroll
            for (int j = 0; j < 4; ++j) {
                const float p = __expf(s[j][r] - mnew);
                s[j][r] = p;
                rs += p;
            }
#pragma unroll
            for (int off = 1; off < 16; off <<= 1)
                rs += __shfl_xor(rs, off, 32);
            rl[r] = rl[r] * al + rs;
#pragma unroll
            for (int j = 0; j < 4; ++j) o[j][r] *= al;
        }

        // ---- P: C-layout -> A-layout via per-wave LDS (same-wave, in-order) ----
        __bf16* pb = &Ps[wave][0];
#pragma unroll
        for (int j = 0; j < 4; ++j)
#pragma unroll
            for (int r = 0; r < 8; ++r)
                pb[(hi * 8 + r) * 64 + j * 16 + l15] = f2bf(s[j][r]);

        BFrag pf[2];
        {
            const __bf16* pp = pb + l15 * 64;
            pf[0].u[0] = *(const uint4*)(pp + hi * 8);
            pf[0].u[1] = *(const uint4*)(pp + 16 + hi * 8);
            pf[1].u[0] = *(const uint4*)(pp + 32 + hi * 8);
            pf[1].u[1] = *(const uint4*)(pp + 48 + hi * 8);
        }

        // ---- O += P * V ----
#pragma unroll
        for (int j = 0; j < 4; ++j) {
            const __bf16* p = &Vt[(j * 16 + l15) * 64 + hi * 8];
            BFrag b0, b1;
            b0.u[0] = *(const uint4*)(p);       b0.u[1] = *(const uint4*)(p + 16);
            b1.u[0] = *(const uint4*)(p + 32);  b1.u[1] = *(const uint4*)(p + 48);
            o[j] = wmma_bf(pf[0].v, b0.v, o[j]);
            o[j] = wmma_bf(pf[1].v, b1.v, o[j]);
        }
        __syncthreads();
    }

    // ---- normalize, merge heads into [B, S, 512] bf16 scratch ----
#pragma unroll
    for (int j = 0; j < 4; ++j) {
#pragma unroll
        for (int r = 0; r < 8; ++r) {
            const int sr  = q0 + hi * 8 + r;
            const int col = h * DKK + j * 16 + l15;
            const float val = o[j][r] / rl[r];
            aw[((size_t)b * SEQ + sr) * HIDDEN + col] = f2bf(val);
        }
    }
}

// ---------------------------------------------------------------------------
extern "C" void kernel_launch(void* const* d_in, const int* in_sizes, int n_in,
                              void* d_out, int out_size, void* d_ws, size_t ws_size,
                              hipStream_t stream)
{
    const float* q         = (const float*)d_in[0];
    const float* k         = (const float*)d_in[1];
    const float* v         = (const float*)d_in[2];
    const float* attn_bias = (const float*)d_in[3];
    const int*   gmask     = (const int*)d_in[4];
    const float* subsq     = (const float*)d_in[5];
    const float* Wq = (const float*)d_in[6];
    const float* bq = (const float*)d_in[7];
    const float* Wk = (const float*)d_in[8];
    const float* bk = (const float*)d_in[9];
    const float* Wv = (const float*)d_in[10];
    const float* bv = (const float*)d_in[11];
    const float* Wo = (const float*)d_in[12];
    const float* bo = (const float*)d_in[13];

    const size_t HE = (size_t)BATCH * HEADS * SEQ * DKK;   // 2M elems each
    __bf16* qh  = (__bf16*)d_ws;
    __bf16* khp = qh + HE;
    __bf16* vhp = khp + HE;
    __bf16* aw  = vhp + HE;                                 // [B,S,512] bf16

    dim3 gg(HIDDEN / 128, (BATCH * SEQ) / 128);             // (4, 32)
    const float scale = 0.125f;                             // 1/sqrt(64)

    gemm_wmma<false, false><<<gg, 256, 0, stream>>>(q, Wq, bq, qh,  scale);
    gemm_wmma<false, false><<<gg, 256, 0, stream>>>(k, Wk, bk, khp, 1.0f);
    gemm_wmma<false, false><<<gg, 256, 0, stream>>>(v, Wv, bv, vhp, 1.0f);

    flash_attn<<<dim3(SEQ / 64, HEADS, BATCH), 128, 0, stream>>>(
        qh, khp, vhp, attn_bias, gmask, subsq, aw);

    gemm_wmma<true, true><<<gg, 256, 0, stream>>>(aw, Wo, bo, d_out, 1.0f);
}
